// Gcn_32023276159196
// MI455X (gfx1250) — matrix operand order repaired and checked by reading, not compiled, not run on hardware
//
#include <hip/hip_runtime.h>
#include <cstdint>
#include <cstddef>

// ---------------------------------------------------------------------------
// GCN on MI455X (gfx1250):
//   per layer: G = X @ W  (bf16-split WMMA, f32 accum)
//              H = scatter-add over edges of val * G[col]
//                  (hardware global_atomic_add_f32, L2-resident working set)
//              X' = relu(H) -> re-split to bf16 hi/lo for the next GEMM
// ---------------------------------------------------------------------------

#define GCN_NFEAT 512
#define GCN_HID   256
#define GCN_NCLS  40

typedef __attribute__((ext_vector_type(8)))  __bf16 v8bf;
typedef __attribute__((ext_vector_type(16))) __bf16 v16bf;
typedef __attribute__((ext_vector_type(8)))  float  v8f;

// Guaranteed single-instruction f32 atomic add (ISA §15.18.3 opcode 86).
// Fire-and-forget: tracked by STOREcnt, drained by the implicit S_WAIT_IDLE
// at S_ENDPGM — no explicit wait needed.
__device__ __forceinline__ void atom_add_f32(float* p, float v) {
    asm volatile("global_atomic_add_f32 %0, %1, off"
                 :
                 : "v"(p), "v"(v)
                 : "memory");
}

// ---------------------------------------------------------------------------
// Pack W [K,M] f32 (row-major) into WMMA B-fragment order, split into bf16
// hi/lo.  Fragment layout for V_WMMA_F32_16X16X32_BF16 B operand (32K x 16):
//   lane l holds column n = nt*16 + (l & 15)
//   element e (0..15) holds K = kt*32 + (l>>4)*8 + (e&7) + (e>>3)*16
// Linear packed index: ((nt*KT + kt)*32 + l)*16 + e
// ---------------------------------------------------------------------------
__global__ void gcn_pack_w(const float* __restrict__ W, int K, int M, int NT,
                           __bf16* __restrict__ ph, __bf16* __restrict__ pl) {
    int KT = K >> 5;
    int total = NT * KT * 512;
    int t = blockIdx.x * blockDim.x + threadIdx.x;
    if (t >= total) return;
    int e    = t & 15;
    int l    = (t >> 4) & 31;
    int tile = t >> 9;
    int kt = tile % KT;
    int nt = tile / KT;
    int n = nt * 16 + (l & 15);
    int k = kt * 32 + ((l >> 4) << 3) + (e & 7) + ((e >> 3) << 4);
    float v = (n < M) ? W[(size_t)k * M + n] : 0.0f;
    __bf16 hi = (__bf16)v;
    __bf16 lo = (__bf16)(v - (float)hi);
    ph[t] = hi;
    pl[t] = lo;
}

// ---------------------------------------------------------------------------
// Split f32 -> bf16 hi/lo (optionally with ReLU first).
// ---------------------------------------------------------------------------
__global__ void gcn_split_bf16(const float* __restrict__ X, size_t n, int relu,
                               __bf16* __restrict__ hi, __bf16* __restrict__ lo) {
    size_t i = (size_t)blockIdx.x * blockDim.x + threadIdx.x;
    if (i >= n) return;
    float v = X[i];
    if (relu) v = v > 0.0f ? v : 0.0f;
    __bf16 h = (__bf16)v;
    hi[i] = h;
    lo[i] = (__bf16)(v - (float)h);
}

// ---------------------------------------------------------------------------
// GEMM: out[N, Mpad] = X[N, K] @ Wpacked  via 16x16x32 bf16 WMMA, 3-term
// split compensation (hi*hi + hi*lo + lo*hi), f32 accumulate.
// One wave computes one 16x16 C tile; 8 waves / 256-thread block share a
// row-tile so their A-fragment loads hit the WGP cache.
// ---------------------------------------------------------------------------
__global__ void gcn_gemm_wmma(const __bf16* __restrict__ xhi,
                              const __bf16* __restrict__ xlo,
                              const __bf16* __restrict__ wph,
                              const __bf16* __restrict__ wpl,
                              float* __restrict__ out,
                              int N, int K, int NT, int Mpad) {
    const int wavesPerBlock = blockDim.x >> 5;
    const int tileId = blockIdx.x * wavesPerBlock + (threadIdx.x >> 5);
    const int lane   = threadIdx.x & 31;
    const int tiles  = NT * (N >> 4);
    if (tileId >= tiles) return;   // wave-uniform: EXEC stays all-1s for WMMA

    const int nt = tileId % NT;
    const int rt = tileId / NT;
    const int KT = K >> 5;

    // A fragment addressing (16-bit A 16x32 layout):
    //   row m = lane&15 ; lane half selects K sub-block of 8
    //   elements [0..7]  = k0 + kg + {0..7}
    //   elements [8..15] = k0 + 16 + kg + {0..7},  kg = (lane>>4)*8
    const int    m  = lane & 15;
    const int    kg = (lane >> 4) << 3;
    const size_t arow = (size_t)(rt * 16 + m) * K + kg;

    const __bf16* bh = wph + (size_t)(nt * KT) * 512 + (size_t)lane * 16;
    const __bf16* bl = wpl + (size_t)(nt * KT) * 512 + (size_t)lane * 16;

    v8f acc = {};
    for (int kt = 0; kt < KT; ++kt) {
        const __bf16* pah = xhi + arow + (size_t)kt * 32;
        const __bf16* pal = xlo + arow + (size_t)kt * 32;
        v8bf ah0 = *(const v8bf*)(pah);
        v8bf ah1 = *(const v8bf*)(pah + 16);
        v8bf al0 = *(const v8bf*)(pal);
        v8bf al1 = *(const v8bf*)(pal + 16);
        v16bf aH = __builtin_shufflevector(ah0, ah1, 0,1,2,3,4,5,6,7,8,9,10,11,12,13,14,15);
        v16bf aL = __builtin_shufflevector(al0, al1, 0,1,2,3,4,5,6,7,8,9,10,11,12,13,14,15);
        v16bf bH = *(const v16bf*)(bh + (size_t)kt * 512);
        v16bf bL = *(const v16bf*)(bl + (size_t)kt * 512);
        acc = __builtin_amdgcn_wmma_f32_16x16x32_bf16(false, aH, false, bH,
                                                      (short)0, acc, false, false);
        acc = __builtin_amdgcn_wmma_f32_16x16x32_bf16(false, aH, false, bL,
                                                      (short)0, acc, false, false);
        acc = __builtin_amdgcn_wmma_f32_16x16x32_bf16(false, aL, false, bH,
                                                      (short)0, acc, false, false);
    }

    // C layout: VGPR r -> row r (lanes 0-15) / row r+8 (lanes 16-31), col = lane&15
    float* crow = out + (size_t)rt * 16 * Mpad + (size_t)nt * 16 + (lane & 15);
    const int roff = (lane >> 4) << 3;
#pragma unroll
    for (int r = 0; r < 8; ++r)
        crow[(size_t)(r + roff) * Mpad] = acc[r];
}

// ---------------------------------------------------------------------------
// SpMM segment-sum: one wave per edge; lanes stride over feature columns.
// Y[row, :] += val * X[col, :].  X rows are L2-resident (102 MB < 192 MB L2).
// Fast path: float4 gathers (global_load_b128) when M % 128 == 0.
// ---------------------------------------------------------------------------
__global__ void gcn_spmm_atomic(const int* __restrict__ rows,
                                const int* __restrict__ cols,
                                const float* __restrict__ vals,
                                const float* __restrict__ X,
                                float* __restrict__ Y,
                                int E, int M, int strideIn, int strideOut) {
    int w = (blockIdx.x * blockDim.x + threadIdx.x) >> 5;
    int lane = threadIdx.x & 31;
    if (w >= E) return;
    int   r = rows[w];
    int   c = cols[w];
    float v = vals[w];
    const float* xp = X + (size_t)c * strideIn;
    float*       yp = Y + (size_t)r * strideOut;
    if ((M & 127) == 0) {
        // vector path: 32 lanes x float4 = 512B contiguous per step
        for (int i = 0; i < M; i += 128) {
            int cc = i + lane * 4;
            float4 xv = *(const float4*)(xp + cc);
            atom_add_f32(yp + cc + 0, v * xv.x);
            atom_add_f32(yp + cc + 1, v * xv.y);
            atom_add_f32(yp + cc + 2, v * xv.z);
            atom_add_f32(yp + cc + 3, v * xv.w);
        }
    } else {
        for (int j = lane; j < M; j += 32)
            atom_add_f32(yp + j, v * xp[j]);
    }
}

__global__ void gcn_relu_inplace(float* __restrict__ p, size_t n) {
    size_t i = (size_t)blockIdx.x * blockDim.x + threadIdx.x;
    if (i >= n) return;
    float v = p[i];
    p[i] = v > 0.0f ? v : 0.0f;
}

// ---------------------------------------------------------------------------
static inline unsigned ceil_div_u(size_t a, unsigned b) {
    return (unsigned)((a + b - 1) / b);
}

extern "C" void kernel_launch(void* const* d_in, const int* in_sizes, int n_in,
                              void* d_out, int out_size, void* d_ws, size_t ws_size,
                              hipStream_t stream) {
    const float* features = (const float*)d_in[0];
    const int*   erow     = (const int*)d_in[1];
    const int*   ecol     = (const int*)d_in[2];
    const float* eval_    = (const float*)d_in[3];
    const float* W_in     = (const float*)d_in[4];
    const float* W_h      = (const float*)d_in[5];
    const float* W_out    = (const float*)d_in[6];

    const int N = in_sizes[0] / GCN_NFEAT;   // 100000 (divisible by 16)
    const int E = in_sizes[1];               // 3.2M

    // ---------------- workspace layout (256B aligned slabs) ----------------
    char* ws = (char*)d_ws;
    size_t off = 0;
    auto alloc = [&](size_t bytes) -> char* {
        char* p = ws + off;
        off += (bytes + 255) & ~(size_t)255;
        return p;
    };
    __bf16* XHI = (__bf16*)alloc((size_t)N * GCN_NFEAT * sizeof(__bf16));
    __bf16* XLO = (__bf16*)alloc((size_t)N * GCN_NFEAT * sizeof(__bf16));
    float*  G   = (float*) alloc((size_t)N * GCN_HID * sizeof(float));
    float*  H   = (float*) alloc((size_t)N * GCN_HID * sizeof(float));
    // packed weights: L0 (KT=16,NT=16)=131072, L1/L2 (KT=8,NT=16)=65536 each,
    // L3 (KT=8,NT=3)=12288  -> 274432 bf16 elements per hi/lo
    __bf16* WPH = (__bf16*)alloc((size_t)274432 * sizeof(__bf16));
    __bf16* WPL = (__bf16*)alloc((size_t)274432 * sizeof(__bf16));
    (void)ws_size;

    __bf16* wph0 = WPH,          * wpl0 = WPL;
    __bf16* wph1 = WPH + 131072, * wpl1 = WPL + 131072;
    __bf16* wph2 = WPH + 196608, * wpl2 = WPL + 196608;
    __bf16* wph3 = WPH + 262144, * wpl3 = WPL + 262144;

    const int T = 256;  // threads per block (8 waves)

    // ---------------- pack weights into WMMA fragment order ----------------
    gcn_pack_w<<<ceil_div_u(131072, T), T, 0, stream>>>(W_in, GCN_NFEAT, GCN_HID, 16, wph0, wpl0);
    gcn_pack_w<<<ceil_div_u(65536,  T), T, 0, stream>>>(W_h,                 GCN_HID, GCN_HID, 16, wph1, wpl1);
    gcn_pack_w<<<ceil_div_u(65536,  T), T, 0, stream>>>(W_h + 65536,         GCN_HID, GCN_HID, 16, wph2, wpl2);
    gcn_pack_w<<<ceil_div_u(12288,  T), T, 0, stream>>>(W_out, GCN_HID, GCN_NCLS, 3, wph3, wpl3);

    const unsigned spmmBlocks = ceil_div_u((size_t)E * 32, T);

    auto run_layer = [&](const float* xin, size_t xin_elems, int relu_in,
                         int K, int NT, int Mpad, int Mout,
                         const __bf16* wh, const __bf16* wl,
                         float* yout, int strideOut) {
        // split (with optional ReLU of previous SpMM output)
        gcn_split_bf16<<<ceil_div_u(xin_elems, T), T, 0, stream>>>(
            xin, xin_elems, relu_in, XHI, XLO);
        // dense GEMM via WMMA
        const int tiles = NT * (N >> 4);
        gcn_gemm_wmma<<<ceil_div_u((size_t)tiles, T / 32), T, 0, stream>>>(
            XHI, XLO, wh, wl, G, N, K, NT, Mpad);
        // zero the aggregation target, then scatter-add over edges
        hipMemsetAsync(yout, 0, (size_t)N * strideOut * sizeof(float), stream);
        gcn_spmm_atomic<<<spmmBlocks, T, 0, stream>>>(
            erow, ecol, eval_, G, yout, E, Mout, Mpad, strideOut);
    };

    // layer 0: features[100k,512] @ W_in -> spmm -> H
    run_layer(features, (size_t)N * GCN_NFEAT, 0, GCN_NFEAT, 16, GCN_HID, GCN_HID,
              wph0, wpl0, H, GCN_HID);
    // layer 1: relu(H) @ W_h[0] -> spmm -> H
    run_layer(H, (size_t)N * GCN_HID, 1, GCN_HID, 16, GCN_HID, GCN_HID,
              wph1, wpl1, H, GCN_HID);
    // layer 2: relu(H) @ W_h[1] -> spmm -> H
    run_layer(H, (size_t)N * GCN_HID, 1, GCN_HID, 16, GCN_HID, GCN_HID,
              wph2, wpl2, H, GCN_HID);
    // output layer: relu(H) @ W_out (Mpad=48, logical 40) -> spmm -> d_out
    run_layer(H, (size_t)N * GCN_HID, 1, GCN_HID, 3, 48, GCN_NCLS,
              wph3, wpl3, (float*)d_out, GCN_NCLS);
    // final ReLU on logits (reference applies activation on the output layer)
    gcn_relu_inplace<<<ceil_div_u((size_t)N * GCN_NCLS, T), T, 0, stream>>>(
        (float*)d_out, (size_t)N * GCN_NCLS);
}
